// CPChannelAttention_103079215457
// MI455X (gfx1250) — compile-verified, matrix operand order checked by reading, not run on hardware
//
#include <hip/hip_runtime.h>
#include <hip/hip_bf16.h>

typedef __attribute__((ext_vector_type(16))) __bf16 v16bf;
typedef __attribute__((ext_vector_type(8)))  float  v8f;

union BF16x16 { v16bf v; unsigned int u[8]; };

#define C_DIM   256
#define P_DIM   49
#define IMG     224
#define PHW     7
#define PITCH1  68          // bf16 elems per FM row in k1 (K padded to 64, +conflict pad)
#define PITCH4  258         // bf16 elems per row in k4 (K = 256, odd-dword stride)
#define N_PATCH 1024

static __device__ __forceinline__ unsigned short f2bf(float f) {
    unsigned int u = __float_as_uint(f);
    unsigned int r = 0x7FFFu + ((u >> 16) & 1u);   // round-to-nearest-even
    return (unsigned short)((u + r) >> 16);
}

// base K of the 2-element pair held in dword j of a v16bf A/B operand (16-bit 16x32 layout)
static __device__ __forceinline__ int wmma_k(int j /*dword 0..7*/, int hf /*lane>>4*/) {
    return ((j >> 2) << 4) + ((j & 3) << 1) + (hf << 3);
}

// ---------------------------------------------------------------------------
// Kernel 1: per-patch Gram via WMMA bf16 -> Sc (softmax) + cov_big
// ---------------------------------------------------------------------------
__global__ __launch_bounds__(256) void k1_gram(const float* __restrict__ x,
                                               float* __restrict__ Sc,
                                               float* __restrict__ cov_big) {
    extern __shared__ char smem[];
    float*          G  = (float*)smem;                                   // 256*256 f32
    unsigned short* FM = (unsigned short*)(smem + 65536 * 4);            // 256*PITCH1 bf16
    float*          MU = (float*)(smem + 65536 * 4 + C_DIM * PITCH1 * 2);// 256 f32

    const int m   = blockIdx.x;
    const int hh  = m >> 5, ww = m & 31;
    const int tid = threadIdx.x;

    MU[tid] = 0.f;
    for (int i = tid; i < C_DIM * (PITCH1 - P_DIM); i += 256) {
        int c = i / (PITCH1 - P_DIM), p = P_DIM + i % (PITCH1 - P_DIM);
        FM[c * PITCH1 + p] = 0;
    }
    __syncthreads();

    const float* xp = x + (size_t)(hh * PHW) * IMG + ww * PHW;
    for (int i = tid; i < C_DIM * PHW; i += 256) {
        int c = i / PHW, hr = i % PHW;
        const float* src = xp + (size_t)c * (IMG * IMG) + hr * IMG;
        float s = 0.f;
#pragma unroll
        for (int wc = 0; wc < PHW; ++wc) {
            float v = src[wc];
            s += v;
            FM[c * PITCH1 + hr * PHW + wc] = f2bf(v);
        }
        atomicAdd(&MU[c], s);
    }
    __syncthreads();

    const int wave = tid >> 5, lane = tid & 31;
    const int hf = lane >> 4, lanelo = lane & 15;

    // each wave owns 2 m-tiles; A operands loaded once, reused across all 16 n-tiles
    for (int tmi = 0; tmi < 2; ++tmi) {
        const int tm = wave * 2 + tmi;
        BF16x16 A0, A1;
#pragma unroll
        for (int j = 0; j < 8; ++j) {
            int kb = wmma_k(j, hf);
            A0.u[j] = *(const unsigned int*)&FM[(tm * 16 + lanelo) * PITCH1 + kb];
            A1.u[j] = *(const unsigned int*)&FM[(tm * 16 + lanelo) * PITCH1 + 32 + kb];
        }
        for (int tn = 0; tn < 16; ++tn) {
            BF16x16 B0, B1;
#pragma unroll
            for (int j = 0; j < 8; ++j) {
                int kb = wmma_k(j, hf);
                B0.u[j] = *(const unsigned int*)&FM[(tn * 16 + lanelo) * PITCH1 + kb];
                B1.u[j] = *(const unsigned int*)&FM[(tn * 16 + lanelo) * PITCH1 + 32 + kb];
            }
            v8f acc = {};
            acc = __builtin_amdgcn_wmma_f32_16x16x32_bf16(false, A0.v, false, B0.v,
                                                          (short)0, acc, false, false);
            acc = __builtin_amdgcn_wmma_f32_16x16x32_bf16(false, A1.v, false, B1.v,
                                                          (short)0, acc, false, false);
#pragma unroll
            for (int r = 0; r < 8; ++r)
                G[(tm * 16 + r + 8 * hf) * 256 + tn * 16 + lanelo] = acc[r];
        }
    }
    __syncthreads();

    // cov_big = G/P - mu*mu^T   (mu = rowsum/P)
    const float invP = 1.f / 49.f;
    float* covp = cov_big + (size_t)m * 65536;
    const float mu_d = MU[tid] * invP;
    for (int k = 0; k < 256; ++k) {
        float mu_c = MU[k] * invP;
        covp[k * 256 + tid] = G[k * 256 + tid] * invP - mu_c * mu_d;
    }

    // softmax rows of G -> Sc
    float* scp = Sc + (size_t)m * 65536;
    for (int row = wave; row < 256; row += 8) {
        float mx = -3.4e38f;
#pragma unroll
        for (int jj = 0; jj < 8; ++jj) mx = fmaxf(mx, G[row * 256 + lane + jj * 32]);
#pragma unroll
        for (int off = 16; off > 0; off >>= 1) mx = fmaxf(mx, __shfl_xor(mx, off, 32));
        float vals[8], sum = 0.f;
#pragma unroll
        for (int jj = 0; jj < 8; ++jj) {
            float e = __expf(G[row * 256 + lane + jj * 32] - mx);
            vals[jj] = e; sum += e;
        }
#pragma unroll
        for (int off = 16; off > 0; off >>= 1) sum += __shfl_xor(sum, off, 32);
        float inv = 1.f / sum;
#pragma unroll
        for (int jj = 0; jj < 8; ++jj) scp[row * 256 + lane + jj * 32] = vals[jj] * inv;
    }
}

// ---------------------------------------------------------------------------
// Kernel 2: cov_global = mean over patches of cov_big (b128 streams + prefetch)
// ---------------------------------------------------------------------------
__global__ __launch_bounds__(256) void k2_covmean(const float* __restrict__ cov_big,
                                                  float* __restrict__ cov_global) {
    const int i4 = blockIdx.x * 256 + threadIdx.x;       // float4 index (grid 64)
    const float4* __restrict__ src = (const float4*)cov_big;
    float sx = 0.f, sy = 0.f, sz = 0.f, sw = 0.f;
    for (int mm = 0; mm < N_PATCH; ++mm) {
        if (mm + 4 < N_PATCH)
            __builtin_prefetch(&src[(size_t)(mm + 4) * 16384 + i4], 0, 0);
        float4 v = src[(size_t)mm * 16384 + i4];
        sx += v.x; sy += v.y; sz += v.z; sw += v.w;
    }
    const float k = 1.f / (float)N_PATCH;
    float4 r; r.x = sx * k; r.y = sy * k; r.z = sz * k; r.w = sw * k;
    ((float4*)cov_global)[i4] = r;
}

// ---------------------------------------------------------------------------
// Kernel 3: CP-ALS (rank 8, 5 sweeps) + cov_cp = (f1*w) @ f2^T
// T staged into LDS with GLOBAL_LOAD_ASYNC_TO_LDS_B128 (ASYNCcnt path)
// ---------------------------------------------------------------------------
static __device__ void invert8(const float* V, float* OUT) {
    float Aq[8][16];
    for (int i = 0; i < 8; ++i)
        for (int j = 0; j < 8; ++j) { Aq[i][j] = V[i * 8 + j]; Aq[i][8 + j] = (i == j) ? 1.f : 0.f; }
    for (int col = 0; col < 8; ++col) {
        int piv = col; float best = fabsf(Aq[col][col]);
        for (int rr = col + 1; rr < 8; ++rr) {
            float v = fabsf(Aq[rr][col]);
            if (v > best) { best = v; piv = rr; }
        }
        if (piv != col)
            for (int jc = 0; jc < 16; ++jc) { float t = Aq[col][jc]; Aq[col][jc] = Aq[piv][jc]; Aq[piv][jc] = t; }
        float d = 1.f / Aq[col][col];
        for (int jc = 0; jc < 16; ++jc) Aq[col][jc] *= d;
        for (int rr = 0; rr < 8; ++rr) if (rr != col) {
            float f = Aq[rr][col];
            for (int jc = 0; jc < 16; ++jc) Aq[rr][jc] -= f * Aq[col][jc];
        }
    }
    for (int i = 0; i < 8; ++i)
        for (int jc = 0; jc < 8; ++jc) OUT[i * 8 + jc] = Aq[i][8 + jc];
}

__global__ __launch_bounds__(256) void k3_parafac(const float* __restrict__ cov_global,
                                                  const float* __restrict__ f0i,
                                                  const float* __restrict__ f1i,
                                                  const float* __restrict__ f2i,
                                                  float* __restrict__ cov_cp) {
    extern __shared__ char smem[];
    float* T    = (float*)smem;       // 65536
    float* f0   = T + 65536;          // 8
    float* f1   = f0 + 8;             // 2048
    float* f2   = f1 + 2048;          // 2048
    float* g0   = f2 + 2048;          // 64
    float* g1   = g0 + 64;            // 64
    float* g2   = g1 + 64;            // 64
    float* Vb   = g2 + 64;            // 64
    float* inv  = Vb + 64;            // 64
    float* mtt  = inv + 64;           // 2048
    float* mtt0 = mtt + 2048;         // 8

    const int tid = threadIdx.x;

    // ---- async DMA: cov_global (256 KB) -> LDS T, 16B per lane per op ----
    {
        unsigned tbase = (unsigned)(size_t)T;   // low 32 bits of flat LDS addr = LDS offset
        for (int it = 0; it < 64; ++it) {
            unsigned byteoff = (unsigned)((it * 256 + tid) * 16);
            unsigned ldsa = tbase + byteoff;
            asm volatile("global_load_async_to_lds_b128 %0, %1, %2"
                         :: "v"(ldsa), "v"(byteoff), "s"(cov_global)
                         : "memory");
        }
#if __has_builtin(__builtin_amdgcn_s_wait_asynccnt)
        __builtin_amdgcn_s_wait_asynccnt(0);
#else
        asm volatile("s_wait_asynccnt 0x0" ::: "memory");
#endif
    }
    if (tid < 8) f0[tid] = f0i[tid];
    for (int i = tid; i < 2048; i += 256) { f1[i] = f1i[i]; f2[i] = f2i[i]; }
    __syncthreads();

    for (int it = 0; it < 5; ++it) {
        // ---- mode 0 : update f0 ----
        if (tid < 64) {
            int r = tid >> 3, s = tid & 7;
            float a1 = 0.f, a2 = 0.f;
            for (int a = 0; a < 256; ++a) {
                a1 += f1[a * 8 + r] * f1[a * 8 + s];
                a2 += f2[a * 8 + r] * f2[a * 8 + s];
            }
            g1[tid] = a1; g2[tid] = a2;
        }
        if (tid < 8) mtt0[tid] = 0.f;
        __syncthreads();
        {
            float acc[8];
#pragma unroll
            for (int r = 0; r < 8; ++r) acc[r] = 0.f;
            for (int b = 0; b < 256; ++b) {
                float tv = T[tid * 256 + b];
#pragma unroll
                for (int r = 0; r < 8; ++r) acc[r] += tv * f2[b * 8 + r];
            }
#pragma unroll
            for (int r = 0; r < 8; ++r) atomicAdd(&mtt0[r], acc[r] * f1[tid * 8 + r]);
        }
        __syncthreads();
        if (tid == 0) { for (int i = 0; i < 64; ++i) Vb[i] = g1[i] * g2[i]; invert8(Vb, inv); }
        __syncthreads();
        if (tid < 8) {
            float s = 0.f;
            for (int ss = 0; ss < 8; ++ss) s += inv[ss * 8 + tid] * mtt0[ss];
            f0[tid] = s;
        }
        __syncthreads();

        // ---- mode 1 : update f1 ----
        if (tid < 64) g0[tid] = f0[tid >> 3] * f0[tid & 7];
        {
            float acc[8];
#pragma unroll
            for (int r = 0; r < 8; ++r) acc[r] = 0.f;
            for (int b = 0; b < 256; ++b) {
                float tv = T[tid * 256 + b];
#pragma unroll
                for (int r = 0; r < 8; ++r) acc[r] += tv * f2[b * 8 + r];
            }
#pragma unroll
            for (int r = 0; r < 8; ++r) mtt[tid * 8 + r] = f0[r] * acc[r];
        }
        __syncthreads();
        if (tid == 0) { for (int i = 0; i < 64; ++i) Vb[i] = g0[i] * g2[i]; invert8(Vb, inv); }
        __syncthreads();
        {
            float nf[8];
#pragma unroll
            for (int r = 0; r < 8; ++r) {
                float s = 0.f;
                for (int ss = 0; ss < 8; ++ss) s += inv[ss * 8 + r] * mtt[tid * 8 + ss];
                nf[r] = s;
            }
#pragma unroll
            for (int r = 0; r < 8; ++r) f1[tid * 8 + r] = nf[r];
        }
        __syncthreads();

        // ---- mode 2 : update f2 ----
        if (tid < 64) {
            int r = tid >> 3, s = tid & 7;
            float a1 = 0.f;
            for (int a = 0; a < 256; ++a) a1 += f1[a * 8 + r] * f1[a * 8 + s];
            g1[tid] = a1;
        }
        {
            float acc[8];
#pragma unroll
            for (int r = 0; r < 8; ++r) acc[r] = 0.f;
            for (int a = 0; a < 256; ++a) {
                float tv = T[a * 256 + tid];
#pragma unroll
                for (int r = 0; r < 8; ++r) acc[r] += tv * f1[a * 8 + r];
            }
#pragma unroll
            for (int r = 0; r < 8; ++r) mtt[tid * 8 + r] = f0[r] * acc[r];
        }
        __syncthreads();
        if (tid == 0) { for (int i = 0; i < 64; ++i) Vb[i] = g0[i] * g1[i]; invert8(Vb, inv); }
        __syncthreads();
        {
            float nf[8];
#pragma unroll
            for (int r = 0; r < 8; ++r) {
                float s = 0.f;
                for (int ss = 0; ss < 8; ++ss) s += inv[ss * 8 + r] * mtt[tid * 8 + ss];
                nf[r] = s;
            }
#pragma unroll
            for (int r = 0; r < 8; ++r) f2[tid * 8 + r] = nf[r];
        }
        __syncthreads();
    }

    // cov_cp[c][d] = sum_r f1[c][r]*f0[r]*f2[d][r]
    for (int k = 0; k < 256; ++k) {
        float s = 0.f;
#pragma unroll
        for (int r = 0; r < 8; ++r) s += f1[k * 8 + r] * f0[r] * f2[tid * 8 + r];
        cov_cp[k * 256 + tid] = s;
    }
}

// ---------------------------------------------------------------------------
// Kernel 4: Ec = (Sc + cov_cp) @ FM via WMMA, fused fold + elementwise finish
// ---------------------------------------------------------------------------
__global__ __launch_bounds__(256) void k4_ec(const float* __restrict__ x,
                                             const float* __restrict__ Sc,
                                             const float* __restrict__ cov_cp,
                                             const float* __restrict__ beta_p,
                                             float* __restrict__ out,
                                             float* __restrict__ ec_map) {
    extern __shared__ char smem[];
    unsigned short* Lb = (unsigned short*)smem;                      // 256 x PITCH4 bf16
    unsigned short* FT = (unsigned short*)(smem + 256 * PITCH4 * 2); // 64 x PITCH4 bf16 (p-major)

    const int m   = blockIdx.x;
    const int hh  = m >> 5, ww = m & 31;
    const int tid = threadIdx.x;
    const float betav = beta_p[0];

    // stage transposed FM tile (FT[p][c]); columns p>=49 never stored -> no zeroing needed
    const float* xp = x + (size_t)(hh * PHW) * IMG + ww * PHW;
    for (int i = tid; i < C_DIM * PHW; i += 256) {
        int c = i / PHW, hr = i % PHW;
        const float* src = xp + (size_t)c * (IMG * IMG) + hr * IMG;
#pragma unroll
        for (int wc = 0; wc < PHW; ++wc)
            FT[(hr * PHW + wc) * PITCH4 + c] = f2bf(src[wc]);
    }
    // stage L = Sc(patch) + cov_cp as bf16
    const float* scp = Sc + (size_t)m * 65536;
    for (int i = tid; i < 65536; i += 256) {
        int c = i >> 8, d = i & 255;
        Lb[c * PITCH4 + d] = f2bf(scp[i] + cov_cp[i]);
    }
    __syncthreads();

    const int wave = tid >> 5, lane = tid & 31;
    const int hf = lane >> 4, lanelo = lane & 15;

    for (int tmi = 0; tmi < 2; ++tmi) {
        int tm = wave + tmi * 8;
        v8f acc[4] = {};
        for (int kc = 0; kc < 8; ++kc) {
            BF16x16 A;
#pragma unroll
            for (int j = 0; j < 8; ++j) {
                int kb = kc * 32 + wmma_k(j, hf);
                A.u[j] = *(const unsigned int*)&Lb[(tm * 16 + lanelo) * PITCH4 + kb];
            }
#pragma unroll
            for (int tn = 0; tn < 4; ++tn) {
                BF16x16 B;
#pragma unroll
                for (int j = 0; j < 8; ++j) {
                    int kb = kc * 32 + wmma_k(j, hf);
                    B.u[j] = *(const unsigned int*)&FT[(tn * 16 + lanelo) * PITCH4 + kb];
                }
                acc[tn] = __builtin_amdgcn_wmma_f32_16x16x32_bf16(false, A.v, false, B.v,
                                                                  (short)0, acc[tn], false, false);
            }
        }
        // epilogue: fold + out = x*(beta*ec + x)
#pragma unroll
        for (int tn = 0; tn < 4; ++tn) {
#pragma unroll
            for (int r = 0; r < 8; ++r) {
                int c = tm * 16 + r + 8 * hf;
                int p = tn * 16 + lanelo;
                if (p < P_DIM) {
                    int h = hh * PHW + p / PHW;
                    int w = ww * PHW + p % PHW;
                    size_t gi = (size_t)c * (IMG * IMG) + (size_t)h * IMG + w;
                    float ec = acc[tn][r];
                    ec_map[gi] = ec;
                    float xv = x[gi];
                    out[gi] = xv * (betav * ec + xv);
                }
            }
        }
    }
}

// ---------------------------------------------------------------------------
extern "C" void kernel_launch(void* const* d_in, const int* in_sizes, int n_in,
                              void* d_out, int out_size, void* d_ws, size_t ws_size,
                              hipStream_t stream) {
    const float* x    = (const float*)d_in[0];
    const float* beta = (const float*)d_in[1];
    const float* f0i  = (const float*)d_in[2];
    const float* f1i  = (const float*)d_in[3];
    const float* f2i  = (const float*)d_in[4];

    float* out     = (float*)d_out;                  // 12,845,056
    float* Sc      = out + 12845056;                 // 67,108,864
    float* cov_big = Sc + 67108864;                  // 67,108,864
    float* ec_map  = cov_big + 67108864;             // 12,845,056

    float* cov_global = (float*)d_ws;                // 65536 f32
    float* cov_cp     = cov_global + 65536;          // 65536 f32

    size_t lds1 = 65536 * 4 + C_DIM * PITCH1 * 2 + C_DIM * 4;   // ~298 KB
    size_t lds3 = (size_t)(65536 + 8 + 2048 + 2048 + 64 * 5 + 2048 + 8) * 4; // ~288 KB
    size_t lds4 = (size_t)(256 + 64) * PITCH4 * 2;              // ~165 KB

    k1_gram<<<N_PATCH, 256, lds1, stream>>>(x, Sc, cov_big);
    k2_covmean<<<64, 256, 0, stream>>>(cov_big, cov_global);
    k3_parafac<<<1, 256, lds3, stream>>>(cov_global, f0i, f1i, f2i, cov_cp);
    k4_ec<<<N_PATCH, 256, lds4, stream>>>(x, Sc, cov_cp, beta, out, ec_map);
}
